// TransformerDenoiser_32719060860968
// MI455X (gfx1250) — matrix operand order repaired and checked by reading, not compiled, not run on hardware
//
#include <hip/hip_runtime.h>
#include <math.h>

// ---------------- problem constants ----------------
constexpr int B_   = 256;
constexpr int L_   = 32;
constexpr int KPK  = 511;      // peaks
constexpr int D_   = 512;
constexpr int H_   = 8;
constexpr int HD_  = 64;       // D/H
constexpr int DFF_ = 2048;
constexpr int NL_  = 6;
constexpr int V_   = 24;
constexpr int S_   = KPK + 1;  // 512 (memory length / bias length)
constexpr int BL_  = B_ * L_;  // 8192 rows

constexpr float PROTON  = 1.007276f;
constexpr float WATER   = 18.010565f;
constexpr float MAX_MZ  = 2000.0f;
constexpr float INV2S2  = 1.0f / (2.0f * 0.1f * 0.1f);

__constant__ float c_res_mass[24] = {
    0.f, 0.f, 0.f,
    57.02146f, 71.03711f, 87.03203f, 97.05276f, 99.06841f, 101.04768f,
    103.00919f, 113.08406f, 113.08406f, 114.04293f, 115.02694f, 128.05858f,
    128.09496f, 129.04259f, 131.04049f, 137.05891f, 147.06841f, 156.10111f,
    163.06333f, 186.07931f, 0.f};

typedef __attribute__((ext_vector_type(16))) _Float16 v16h;
typedef __attribute__((ext_vector_type(4)))  _Float16 h4;
typedef __attribute__((ext_vector_type(8)))  float    v8f;

// ---- Tensor Data Mover availability --------------------------------------
// Both toolchains declare the builtin, but with different arity:
//   ROCm 7.2 (clang-22):      5 args (g0, g1, g2, g3, cpol)
//   amdgpu-toolchain (>=23):  6 args (g0, g1, g2, g3, g4, cpol)
#if defined(__AMDGCN__) &&                                      \
    __has_builtin(__builtin_amdgcn_tensor_load_to_lds) &&       \
    __has_builtin(__builtin_amdgcn_s_wait_tensorcnt)
#define USE_TDM 1
#else
#define USE_TDM 0
#endif

#if USE_TDM
typedef __attribute__((ext_vector_type(4))) unsigned int u32x4;
typedef __attribute__((ext_vector_type(8))) int          i32x8;
typedef __attribute__((ext_vector_type(4))) int          i32x4;

// Issue a TDM load of a 2D tile (tile_w x tile_h elements, 4-byte elements,
// row stride row_stride elements) from global memory into LDS at lds_off.
// D# layout per CDNA5 ISA.
__device__ inline void tdm_load_2d_f32(unsigned lds_off, const void* gaddr,
                                       unsigned tile_w, unsigned tile_h,
                                       unsigned row_stride)
{
  const unsigned long long ga = (unsigned long long)gaddr;
  u32x4 g0;
  g0[0] = 1u;                                            // count=1 (valid, user)
  g0[1] = lds_off;                                       // lds_addr (bytes)
  g0[2] = (unsigned)(ga & 0xFFFFFFFFu);                  // global_addr lo
  g0[3] = (unsigned)((ga >> 32) & 0x1FFFFFFu) | (2u << 30); // addr hi | type=2
  i32x8 g1;
  g1[0] = (int)(2u << 16);                               // data_size=2 -> 4B
  g1[1] = (int)((row_stride & 0xFFFFu) << 16);           // tensor_dim0[15:0]@48
  g1[2] = (int)(((row_stride >> 16) & 0xFFFFu) |
                (0x8000u << 16));                        // dim0 hi | tensor_dim1 lo (32768)
  g1[3] = (int)((tile_w & 0xFFFFu) << 16);               // tensor_dim1 hi | tile_dim0
  g1[4] = (int)(tile_h & 0xFFFFu);                       // tile_dim1 | tile_dim2=0
  g1[5] = (int)row_stride;                               // tensor_dim0_stride lo32
  g1[6] = 0;                                             // stride0 hi | stride1 lo
  g1[7] = 0;                                             // stride1 hi
  i32x4 z4 = {0, 0, 0, 0};                               // groups 2/3 unused (2D)
#if defined(__clang_major__) && (__clang_major__ >= 23)
  i32x8 z8 = {0, 0, 0, 0, 0, 0, 0, 0};
  __builtin_amdgcn_tensor_load_to_lds(g0, g1, z4, z4, z8, 0);
#else
  __builtin_amdgcn_tensor_load_to_lds(g0, g1, z4, z4, 0);
#endif
}
#endif

// ---------------------------------------------------------------------------
// GEMM: C[M][N] = act(A[M][Kd] @ W[N][Kd]^T + bias) + R
// Block = 128 threads (4 waves). Block tile 64x64, K-step 32.
// Each wave owns a 2x2 grid of 16x16 WMMA tiles (4 v_wmma per K-step).
// LDS holds tiles pre-swizzled into WMMA *fragment layout* so a fragment load
// is one aligned 32-byte LDS vector read (2x ds_load_b128), no gathers.
//
// Fragment layouts (CDNA5 ISA):
//  A 16x32 f16: element (m,k) -> lane = m + 16*((k>>3)&1),
//                               idx  = (k&7) + 8*((k>>4)&1)
//  B 32x16 f16 (from W rows): element (n,k) -> lane = n + 16*((k>>3)&1),
//                                              idx  = (k&7) + 8*((k>>4)&1)
// ---------------------------------------------------------------------------
template <bool NG>
__global__ void __launch_bounds__(128)
k_gemm(const float* __restrict__ A, const float* __restrict__ W,
       const float* __restrict__ bias, const float* __restrict__ R,
       float* __restrict__ C, int M, int N, int Kd, int act)
{
  __shared__ v16h sA[4 * 32];   // 4 m-subtiles x 32 lanes
  __shared__ v16h sW[4 * 32];   // 4 n-subtiles x 32 lanes
  const int tid  = threadIdx.x;
  const int lane = tid & 31;
  const int wave = tid >> 5;
  const int wr   = wave >> 1;     // wave row (0/1) -> m-subtiles 2*wr,2*wr+1
  const int wc   = wave & 1;      // wave col (0/1) -> n-subtiles 2*wc,2*wc+1
  const int m0   = blockIdx.y * 64;
  const int n0   = blockIdx.x * 64;

  v8f acc[2][2] = {};

  for (int k0 = 0; k0 < Kd; k0 += 32) {
    // ---- load 64x32 f32 A-tile and W-tile: 512 float4 each, 4/thread ----
    float4 av[4], wv[4];
#pragma unroll
    for (int f = 0; f < 4; ++f) {
      const int id  = tid + 128 * f;        // 0..511
      const int row = id >> 3;              // 0..63
      const int k4  = (id & 7) << 2;        // 0,4,...,28
      av[f] = *(const float4*)&A[(size_t)(m0 + row) * Kd + (k0 + k4)];
      if (NG) {
        wv[f] = make_float4(0.f, 0.f, 0.f, 0.f);
        if (n0 + row < N)
          wv[f] = *(const float4*)&W[(size_t)(n0 + row) * Kd + (k0 + k4)];
      } else {
        wv[f] = *(const float4*)&W[(size_t)(n0 + row) * Kd + (k0 + k4)];
      }
    }
    if (k0 + 32 < Kd && tid < 64) {
      __builtin_prefetch(&A[(size_t)(m0 + tid) * Kd + (k0 + 32)], 0, 1);
      if (!NG || (n0 + tid < N))
        __builtin_prefetch(&W[(size_t)(n0 + tid) * Kd + (k0 + 32)], 0, 1);
    }
    __syncthreads();
    // ---- convert + store into fragment-layout LDS (one b64 store each) ----
#pragma unroll
    for (int f = 0; f < 4; ++f) {
      const int id   = tid + 128 * f;
      const int row  = id >> 3;
      const int k4   = (id & 7) << 2;
      h4 ah; ah[0] = (_Float16)av[f].x; ah[1] = (_Float16)av[f].y;
             ah[2] = (_Float16)av[f].z; ah[3] = (_Float16)av[f].w;
      h4 wh; wh[0] = (_Float16)wv[f].x; wh[1] = (_Float16)wv[f].y;
             wh[2] = (_Float16)wv[f].z; wh[3] = (_Float16)wv[f].w;
      // A fragment position
      const int at   = row >> 4;                       // m-subtile
      const int am   = row & 15;
      const int alan = am + (((k4 >> 3) & 1) << 4);
      const int aidx = (k4 & 7) + ((k4 >> 4) << 3);
      *(h4*)((_Float16*)&sA[at * 32 + alan] + aidx) = ah;
      // B fragment position
      const int wt   = row >> 4;                       // n-subtile
      const int wn   = row & 15;
      const int wlan = wn + (((k4 >> 3) & 1) << 4);
      const int widx = (k4 & 7) + ((k4 >> 4) << 3);
      *(h4*)((_Float16*)&sW[wt * 32 + wlan] + widx) = wh;
    }
    __syncthreads();
    // ---- 4 WMMAs per wave ----
    const v16h a0 = sA[(2 * wr + 0) * 32 + lane];
    const v16h a1 = sA[(2 * wr + 1) * 32 + lane];
    const v16h b0 = sW[(2 * wc + 0) * 32 + lane];
    const v16h b1 = sW[(2 * wc + 1) * 32 + lane];
    acc[0][0] = __builtin_amdgcn_wmma_f32_16x16x32_f16(false, a0, false, b0,
                                                       (short)0, acc[0][0], false, false);
    acc[0][1] = __builtin_amdgcn_wmma_f32_16x16x32_f16(false, a0, false, b1,
                                                       (short)0, acc[0][1], false, false);
    acc[1][0] = __builtin_amdgcn_wmma_f32_16x16x32_f16(false, a1, false, b0,
                                                       (short)0, acc[1][0], false, false);
    acc[1][1] = __builtin_amdgcn_wmma_f32_16x16x32_f16(false, a1, false, b1,
                                                       (short)0, acc[1][1], false, false);
  }

  // ---- epilogue: C/D layout lane 0-15 -> M=r, lane 16-31 -> M=8+r ----
  const int nn = lane & 15;
  const int mb = (lane < 16) ? 0 : 8;
#pragma unroll
  for (int i = 0; i < 2; ++i) {
#pragma unroll
    for (int j = 0; j < 2; ++j) {
      const int n = n0 + (2 * wc + j) * 16 + nn;
      if (NG && n >= N) continue;
      const float bn = bias ? bias[n] : 0.0f;
#pragma unroll
      for (int r = 0; r < 8; ++r) {
        const size_t mi = (size_t)(m0 + (2 * wr + i) * 16 + mb + r);
        float v = acc[i][j][r] + bn;
        if (act == 1) v = 0.5f * v * (1.0f + erff(v * 0.70710678118f));
        if (R) v += R[mi * N + n];
        C[mi * N + n] = v;
      }
    }
  }
}

// ---------------- LayerNorm: one wave per row of D=512 ----------------
__global__ void __launch_bounds__(32)
k_ln(const float* __restrict__ x, const float* __restrict__ g,
     const float* __restrict__ be, float* __restrict__ y)
{
  const size_t row = blockIdx.x;
  const int lane = threadIdx.x;
  const float* xr = x + row * D_;
  float v[16];
  float s = 0.f;
#pragma unroll
  for (int j = 0; j < 16; ++j) { v[j] = xr[lane + 32 * j]; s += v[j]; }
#pragma unroll
  for (int off = 16; off >= 1; off >>= 1) s += __shfl_xor(s, off, 32);
  const float mean = s * (1.0f / D_);
  float q = 0.f;
#pragma unroll
  for (int j = 0; j < 16; ++j) { const float d0 = v[j] - mean; q += d0 * d0; }
#pragma unroll
  for (int off = 16; off >= 1; off >>= 1) q += __shfl_xor(q, off, 32);
  const float inv = rsqrtf(q * (1.0f / D_) + 1e-5f);
  float* yr = y + row * D_;
#pragma unroll
  for (int j = 0; j < 16; ++j) {
    const int d0 = lane + 32 * j;
    yr[d0] = (v[j] - mean) * inv * g[d0] + be[d0];
  }
}

// ---------------- time embedding: temb[b] = sincos(t*freq) @ time_w^T + b ---
__global__ void __launch_bounds__(256)
k_time_emb(const int* __restrict__ t, const float* __restrict__ time_w,
           const float* __restrict__ time_b, float* __restrict__ temb)
{
  __shared__ float sc[D_];
  const int b = blockIdx.x, tid = threadIdx.x;
  const float tv = (float)t[b];
  const float fr = expf(-logf(10000.0f) * (float)tid / 256.0f);
  const float tf = tv * fr;
  sc[tid]       = sinf(tf);
  sc[tid + 256] = cosf(tf);
  __syncthreads();
#pragma unroll
  for (int rep = 0; rep < 2; ++rep) {
    const int d0 = tid + 256 * rep;
    const float* wr = time_w + (size_t)d0 * D_;
    float acc = time_b[d0];
    for (int j = 0; j < D_; ++j) acc += sc[j] * wr[j];
    temb[(size_t)b * D_ + d0] = acc;
  }
}

// ---------------- x = tok_emb[xt] + pos_emb + temb ----------------
__global__ void __launch_bounds__(256)
k_embed(const int* __restrict__ xt, const float* __restrict__ tok_emb,
        const float* __restrict__ pos_emb, const float* __restrict__ temb,
        float* __restrict__ x)
{
  const int bl = blockIdx.x;           // b*L + l
  const int b = bl / L_, l = bl % L_;
  const int tok = xt[bl];
#pragma unroll
  for (int rep = 0; rep < 2; ++rep) {
    const int d0 = threadIdx.x + 256 * rep;
    x[(size_t)bl * D_ + d0] = tok_emb[(size_t)tok * D_ + d0]
                            + pos_emb[(size_t)l * D_ + d0]
                            + temb[(size_t)b * D_ + d0];
  }
}

// ---------------- b/y-ion Gaussian bias: bb[b][l][0]=0, bb[b][l][1+k] -------
__global__ void __launch_bounds__(256)
k_by_bias(const int* __restrict__ xt, const float* __restrict__ peaks,
          const float* __restrict__ pm, float* __restrict__ bb)
{
  const int bl = blockIdx.x;
  const int b = bl / L_, l = bl % L_;
  float cum = 0.f;
  for (int j = 0; j <= l; ++j) {
    const int tk = xt[b * L_ + j];
    cum += (tk >= 3 && tk < 23) ? c_res_mass[tk] : 0.f;
  }
  const int tok = xt[bl];
  const float is_aa = (tok >= 3 && tok < 23) ? 1.0f : 0.0f;
  const float pmb = pm[b];
  const float valid = (pmb > 1.0f) ? 1.0f : 0.0f;
  const float b_ion = cum + PROTON;
  const float y_ion = (pmb - cum + WATER + PROTON) * valid;
  if (threadIdx.x == 0) bb[(size_t)bl * S_] = 0.f;
  for (int k = threadIdx.x; k < KPK; k += 256) {
    const float mz    = peaks[((size_t)b * KPK + k) * 2] * MAX_MZ;
    const float inten = peaks[((size_t)b * KPK + k) * 2 + 1];
    const float bd = b_ion - mz, yd = y_ion - mz;
    const float prox = (expf(-bd * bd * INV2S2) + expf(-yd * yd * INV2S2))
                       * inten * is_aa;
    bb[(size_t)bl * S_ + 1 + k] = prox;
  }
}

// ---------------- fused self-attention: one wave per (b,h), lane = query l --
// K/V tiles (32x64 f32, row stride 3*D) staged to LDS via the Tensor Data
// Mover when available (TENSORcnt-tracked, EXEC-independent), else vector
// loads.
__global__ void __launch_bounds__(32)
k_sa_attn(const float* __restrict__ qkv, float* __restrict__ out)
{
  __shared__ float sK[L_][HD_];
  __shared__ float sV[L_][HD_];
  const int b = blockIdx.x / H_, h = blockIdx.x % H_;
  const int lane = threadIdx.x;
#if USE_TDM
  {
    const float* kbase = qkv + (size_t)(b * L_) * (3 * D_) + D_ + h * HD_;
    const float* vbase = qkv + (size_t)(b * L_) * (3 * D_) + 2 * D_ + h * HD_;
    tdm_load_2d_f32((unsigned)(size_t)&sK[0][0], kbase, HD_, L_, 3 * D_);
    tdm_load_2d_f32((unsigned)(size_t)&sV[0][0], vbase, HD_, L_, 3 * D_);
    __builtin_amdgcn_s_wait_tensorcnt(0);
  }
#else
#pragma unroll
  for (int j = 0; j < 64; ++j) {         // stage K,V (32x64 each)
    const int i = lane + 32 * j;
    const int s = i >> 6, d = i & 63;
    const size_t base = (size_t)(b * L_ + s) * (3 * D_) + h * HD_ + d;
    sK[s][d] = qkv[base + D_];
    sV[s][d] = qkv[base + 2 * D_];
  }
#endif
  __syncthreads();
  float q[HD_];
  const float scale = 0.125f;            // 1/sqrt(64)
#pragma unroll
  for (int d = 0; d < HD_; ++d)
    q[d] = qkv[(size_t)(b * L_ + lane) * (3 * D_) + h * HD_ + d] * scale;
  float lg[L_];
  float mx = -1e30f;
#pragma unroll
  for (int s = 0; s < L_; ++s) {
    float acc = 0.f;
#pragma unroll
    for (int d = 0; d < HD_; ++d) acc += q[d] * sK[s][d];
    lg[s] = acc;
    mx = fmaxf(mx, acc);
  }
  float sum = 0.f;
#pragma unroll
  for (int s = 0; s < L_; ++s) { lg[s] = expf(lg[s] - mx); sum += lg[s]; }
  const float inv = 1.0f / sum;
  float o[HD_];
#pragma unroll
  for (int d = 0; d < HD_; ++d) o[d] = 0.f;
#pragma unroll
  for (int s = 0; s < L_; ++s) {
    const float p = lg[s] * inv;
#pragma unroll
    for (int d = 0; d < HD_; ++d) o[d] += p * sV[s][d];
  }
#pragma unroll
  for (int d = 0; d < HD_; ++d)
    out[(size_t)(b * L_ + lane) * D_ + h * HD_ + d] = o[d];
}

// ---------------- cross-attn logits + softmax: one wave per (b,h,l) --------
__global__ void __launch_bounds__(32)
k_ca_attn_softmax(const float* __restrict__ q, const float* __restrict__ kmem,
                  const float* __restrict__ bb, const float* __restrict__ bias_scale,
                  int layer, float* __restrict__ probs)
{
  const int bid = blockIdx.x;            // B*H*L
  const int l = bid % L_;
  const int h = (bid / L_) % H_;
  const int b = bid / (L_ * H_);
  const int lane = threadIdx.x;
  const float bsc = bias_scale[layer];
  float qr[HD_];
#pragma unroll
  for (int d = 0; d < HD_; ++d)
    qr[d] = q[(size_t)(b * L_ + l) * D_ + h * HD_ + d] * 0.125f;
  float lg[16];
  float mx = -1e30f;
#pragma unroll
  for (int j = 0; j < 16; ++j) {
    const int s = lane + 32 * j;
    const float* kr = kmem + (size_t)(b * S_ + s) * D_ + h * HD_;
    float acc = 0.f;
#pragma unroll
    for (int d = 0; d < HD_; ++d) acc += qr[d] * kr[d];
    acc += bsc * bb[(size_t)(b * L_ + l) * S_ + s];
    lg[j] = acc;
    mx = fmaxf(mx, acc);
  }
#pragma unroll
  for (int off = 16; off >= 1; off >>= 1) mx = fmaxf(mx, __shfl_xor(mx, off, 32));
  float sum = 0.f;
#pragma unroll
  for (int j = 0; j < 16; ++j) { lg[j] = expf(lg[j] - mx); sum += lg[j]; }
#pragma unroll
  for (int off = 16; off >= 1; off >>= 1) sum += __shfl_xor(sum, off, 32);
  const float inv = 1.0f / sum;
#pragma unroll
  for (int j = 0; j < 16; ++j) {
    const int s = lane + 32 * j;
    probs[(((size_t)(b * H_ + h)) * L_ + l) * S_ + s] = lg[j] * inv;
  }
}

// ---------------- cross-attn PV: block per (b,h); thread owns (l,d) lanes ---
__global__ void __launch_bounds__(256)
k_ca_out(const float* __restrict__ probs, const float* __restrict__ vmem,
         float* __restrict__ out)
{
  const int b = blockIdx.x / H_, h = blockIdx.x % H_;
  const int t = threadIdx.x;
  const int d  = t & 63;
  const int l0 = t >> 6;                 // 0..3
#pragma unroll
  for (int li = 0; li < 8; ++li) {
    const int l = l0 + 4 * li;
    float acc = 0.f;
    const float* p = probs + (((size_t)(b * H_ + h)) * L_ + l) * S_;
    const float* v = vmem + (size_t)b * S_ * D_ + h * HD_ + d;
    for (int s = 0; s < S_; ++s) acc += p[s] * v[(size_t)s * D_];
    out[(size_t)(b * L_ + l) * D_ + h * HD_ + d] = acc;
  }
}

// ---------------- host driver ----------------
extern "C" void kernel_launch(void* const* d_in, const int* in_sizes, int n_in,
                              void* d_out, int out_size, void* d_ws, size_t ws_size,
                              hipStream_t stream)
{
  (void)in_sizes; (void)n_in; (void)out_size; (void)ws_size;
  const int*   xt         = (const int*)  d_in[0];
  const int*   t          = (const int*)  d_in[1];
  const float* peaks      = (const float*)d_in[2];
  const float* pmass      = (const float*)d_in[3];
  const float* memory     = (const float*)d_in[4];
  const float* tok_emb    = (const float*)d_in[5];
  const float* pos_emb    = (const float*)d_in[6];
  const float* time_w     = (const float*)d_in[7];
  const float* time_b     = (const float*)d_in[8];
  const float* sa_qkv_w   = (const float*)d_in[9];
  const float* sa_qkv_b   = (const float*)d_in[10];
  const float* sa_out_w   = (const float*)d_in[11];
  const float* sa_out_b   = (const float*)d_in[12];
  const float* ca_qkv_w   = (const float*)d_in[13];
  const float* ca_qkv_b   = (const float*)d_in[14];
  const float* ca_out_w   = (const float*)d_in[15];
  const float* ca_out_b   = (const float*)d_in[16];
  const float* ff_w1      = (const float*)d_in[17];
  const float* ff_b1      = (const float*)d_in[18];
  const float* ff_w2      = (const float*)d_in[19];
  const float* ff_b2      = (const float*)d_in[20];
  const float* ln1_g      = (const float*)d_in[21];
  const float* ln1_b      = (const float*)d_in[22];
  const float* ln2_g      = (const float*)d_in[23];
  const float* ln2_b      = (const float*)d_in[24];
  const float* ln3_g      = (const float*)d_in[25];
  const float* ln3_b      = (const float*)d_in[26];
  const float* bias_scale = (const float*)d_in[27];
  const float* fnorm_g    = (const float*)d_in[28];
  const float* fnorm_b    = (const float*)d_in[29];
  const float* out_w      = (const float*)d_in[30];
  const float* out_b      = (const float*)d_in[31];

  float* ws = (float*)d_ws;
  size_t off = 0;
  auto alloc = [&](size_t n) { float* p = ws + off; off += n; return p; };
  float* x     = alloc((size_t)BL_ * D_);
  float* h     = alloc((size_t)BL_ * D_);
  float* temb  = alloc((size_t)B_ * D_);
  float* bb    = alloc((size_t)BL_ * S_);
  float* qkv   = alloc((size_t)BL_ * 3 * D_);
  float* attn  = alloc((size_t)BL_ * D_);
  float* ffh   = alloc((size_t)BL_ * DFF_);
  float* kmem  = alloc((size_t)B_ * S_ * D_);
  float* vmem  = alloc((size_t)B_ * S_ * D_);
  float* probs = alloc((size_t)B_ * H_ * L_ * S_);

  k_time_emb<<<B_,  256, 0, stream>>>(t, time_w, time_b, temb);
  k_embed   <<<BL_, 256, 0, stream>>>(xt, tok_emb, pos_emb, temb, x);
  k_by_bias <<<BL_, 256, 0, stream>>>(xt, peaks, pmass, bb);

  for (int i = 0; i < NL_; ++i) {
    const size_t wq3 = (size_t)i * 3 * D_ * D_;
    // ---- self-attention ----
    k_ln<<<BL_, 32, 0, stream>>>(x, ln1_g + i * D_, ln1_b + i * D_, h);
    k_gemm<false><<<dim3(3 * D_ / 64, BL_ / 64), 128, 0, stream>>>(
        h, sa_qkv_w + wq3, sa_qkv_b + (size_t)i * 3 * D_, nullptr, qkv,
        BL_, 3 * D_, D_, 0);
    k_sa_attn<<<B_ * H_, 32, 0, stream>>>(qkv, attn);
    k_gemm<false><<<dim3(D_ / 64, BL_ / 64), 128, 0, stream>>>(
        attn, sa_out_w + (size_t)i * D_ * D_, sa_out_b + (size_t)i * D_, x, x,
        BL_, D_, D_, 0);
    // ---- cross-attention ----
    k_ln<<<BL_, 32, 0, stream>>>(x, ln2_g + i * D_, ln2_b + i * D_, h);
    k_gemm<false><<<dim3(D_ / 64, BL_ / 64), 128, 0, stream>>>(
        h, ca_qkv_w + wq3, ca_qkv_b + (size_t)i * 3 * D_, nullptr, qkv,
        BL_, D_, D_, 0);                       // Q (first D rows of qkv_w)
    k_gemm<false><<<dim3(D_ / 64, (B_ * S_) / 64), 128, 0, stream>>>(
        memory, ca_qkv_w + wq3 + (size_t)D_ * D_,
        ca_qkv_b + (size_t)i * 3 * D_ + D_, nullptr, kmem,
        B_ * S_, D_, D_, 0);                   // K over memory
    k_gemm<false><<<dim3(D_ / 64, (B_ * S_) / 64), 128, 0, stream>>>(
        memory, ca_qkv_w + wq3 + (size_t)2 * D_ * D_,
        ca_qkv_b + (size_t)i * 3 * D_ + 2 * D_, nullptr, vmem,
        B_ * S_, D_, D_, 0);                   // V over memory
    k_ca_attn_softmax<<<B_ * H_ * L_, 32, 0, stream>>>(
        qkv, kmem, bb, bias_scale, i, probs);
    k_ca_out<<<B_ * H_, 256, 0, stream>>>(probs, vmem, attn);
    k_gemm<false><<<dim3(D_ / 64, BL_ / 64), 128, 0, stream>>>(
        attn, ca_out_w + (size_t)i * D_ * D_, ca_out_b + (size_t)i * D_, x, x,
        BL_, D_, D_, 0);
    // ---- FFN ----
    k_ln<<<BL_, 32, 0, stream>>>(x, ln3_g + i * D_, ln3_b + i * D_, h);
    k_gemm<false><<<dim3(DFF_ / 64, BL_ / 64), 128, 0, stream>>>(
        h, ff_w1 + (size_t)i * DFF_ * D_, ff_b1 + (size_t)i * DFF_, nullptr, ffh,
        BL_, DFF_, D_, 1);                     // exact GELU fused
    k_gemm<false><<<dim3(D_ / 64, BL_ / 64), 128, 0, stream>>>(
        ffh, ff_w2 + (size_t)i * D_ * DFF_, ff_b2 + (size_t)i * D_, x, x,
        BL_, D_, DFF_, 0);
  }

  // ---- final norm + vocab projection (N=24 tail, guarded variant) ----
  k_ln<<<BL_, 32, 0, stream>>>(x, fnorm_g, fnorm_b, h);
  k_gemm<true><<<dim3(1, BL_ / 64), 128, 0, stream>>>(
      h, out_w, out_b, nullptr, (float*)d_out, BL_, V_, D_, 0);
}
